// QuantizedHyperbolicAttention_74053826118128
// MI455X (gfx1250) — matrix operand order repaired and checked by reading, not compiled, not run on hardware
//
#include <hip/hip_runtime.h>
#include <hip/hip_bf16.h>

// ---------------------------------------------------------------------------
// MHA forward for MI455X (gfx1250, wave32, WMMA f32_16x16x32_f16).
// ~69 GFLOP vs ~50MB compulsory HBM traffic -> compute bound, so every matmul
// stage runs on the WMMA pipe in f16 with f32 accumulation.
//  * GEMMs: 64x64 register-blocked tile per wave (16 C-fragments) -> ~16
//    flop/byte from L2 instead of 4, so the WMMA pipe (not L2) is the limiter.
//  * Attention: flash-style single pass; the 4 waves of a block share (b,h)
//    and cooperatively stage K (row-major) + V (pre-transposed) in LDS, so
//    all fragments are contiguous 16B ds/global loads.
//  * global_prefetch_b8 on next tiles via __builtin_prefetch.
// ---------------------------------------------------------------------------

typedef __attribute__((ext_vector_type(16))) _Float16 v16h;
typedef __attribute__((ext_vector_type(8)))  _Float16 v8h;
typedef __attribute__((ext_vector_type(8)))  float    v8f;
typedef __attribute__((ext_vector_type(4)))  int      v4i;

#define D_MODEL 1024
#define NHEAD   16
#define HD      64
#define SEQ     2048
#define BATCH   2
#define TOKENS  (BATCH * SEQ)          // 4096

// ---- WMMA fragment helpers (contiguous k order, same for A and B, wave32) ----------
// A (16x32, 16-bit): lane<16 holds row M=l, halves 0..15 -> K=kb+0..15;
//                    lanes 16..31 -> K=kb+16..31.
static __device__ __forceinline__
v16h load_a_frag(const _Float16* base, int ld, int row, int kb, int hi) {
    return *(const v16h*)(base + (size_t)row * ld + kb + hi * 16);
}

// B (32x16, 16-bit): lane<16 holds col N=l, halves 0..15 -> K=kb+0..15;
//                    lanes 16..31 -> K=kb+16..31.  B[k][n] = W[n][k], so this
//                    is a contiguous 32-byte load along W's row n.
static __device__ __forceinline__
v16h load_b_frag(const _Float16* base, int ld, int col, int kb, int hi) {
    return *(const v16h*)(base + (size_t)col * ld + kb + hi * 16);
}

static __device__ __forceinline__
v8f wmma_f16(v16h a, v16h b, v8f c) {
    return __builtin_amdgcn_wmma_f32_16x16x32_f16(
        /*neg_a=*/false, a, /*neg_b=*/false, b,
        /*c_mod=*/(short)0, c, /*reuse_a=*/false, /*reuse_b=*/false);
}

// ---------------------------------------------------------------------------
// 1) fp32 -> fp16 conversion
// ---------------------------------------------------------------------------
__global__ void f32_to_f16_kernel(const float* __restrict__ in,
                                  _Float16* __restrict__ out, int n) {
    for (int i = blockIdx.x * blockDim.x + threadIdx.x; i < n;
         i += gridDim.x * blockDim.x)
        out[i] = (_Float16)in[i];
}

// ---------------------------------------------------------------------------
// 2) Fused QKV projection: [4096 x 1024] x (Wq|Wk|Wv)^T -> q,k,v in f16
//    stored as [mat][B][H][SEQ][HD].
//    One wave computes a 64x64 tile: 4 A-frags x 4 B-frags -> 16 WMMAs/k-step.
// ---------------------------------------------------------------------------
__global__ __launch_bounds__(128)
void qkv_gemm_kernel(const _Float16* __restrict__ xh,
                     const _Float16* __restrict__ wq,
                     const _Float16* __restrict__ wk,
                     const _Float16* __restrict__ wv,
                     _Float16* __restrict__ qkv) {
    const int wave = threadIdx.x >> 5;
    const int lane = threadIdx.x & 31;
    const int l  = lane & 15;
    const int hi = lane >> 4;

    const int NT   = (3 * D_MODEL) / 64;          // 48 col tiles
    const int tile = blockIdx.x * 4 + wave;       // 0..3071
    const int tm   = tile / NT;                   // 0..63
    const int tn   = tile % NT;                   // 0..47
    const int row0 = tm * 64;
    const int col0 = tn * 64;
    const int mat  = col0 / D_MODEL;              // 0:q 1:k 2:v
    const int oo0  = col0 % D_MODEL;              // 64-aligned -> one head

    const _Float16* W = (mat == 0) ? wq : ((mat == 1) ? wk : wv);

    v8f c[4][4] = {};
    for (int k = 0; k < D_MODEL; k += 32) {
        v16h a[4], b[4];
#pragma unroll
        for (int i = 0; i < 4; i++)
            a[i] = load_a_frag(xh, D_MODEL, row0 + i * 16 + l, k, hi);
#pragma unroll
        for (int j = 0; j < 4; j++)
            b[j] = load_b_frag(W, D_MODEL, oo0 + j * 16 + l, k, hi);
        if (k + 128 < D_MODEL) {                  // prefetch next k-block
            __builtin_prefetch(xh + (size_t)(row0 + l) * D_MODEL + k + 128, 0, 3);
            __builtin_prefetch(W  + (size_t)(oo0  + l) * D_MODEL + k + 128, 0, 3);
        }
#pragma unroll
        for (int i = 0; i < 4; i++)
#pragma unroll
            for (int j = 0; j < 4; j++)
                c[i][j] = wmma_f16(a[i], b[j], c[i][j]);
    }

    // C layout: lane holds col n=l; VGPR v holds row m=v+8*hi.
    const int head = oo0 / HD;
#pragma unroll
    for (int i = 0; i < 4; i++) {
#pragma unroll
        for (int j = 0; j < 4; j++) {
            const int dd = j * 16 + l;            // 0..63 within head
#pragma unroll
            for (int v = 0; v < 8; v++) {
                const int t  = row0 + i * 16 + v + 8 * hi;   // token index
                const int bb = t / SEQ, nn = t % SEQ;
                const size_t idx =
                    ((((size_t)mat * BATCH + bb) * NHEAD + head) * SEQ + nn) * HD + dd;
                qkv[idx] = (_Float16)c[i][j][v];
            }
        }
    }
}

// ---------------------------------------------------------------------------
// 3) Flash attention. 4 waves/block share one (b,h); each wave owns a
//    16-query tile.  Per 32-key step the block cooperatively stages:
//      ldsK [32 keys][64 dd]  (row-major, for S^T = K @ Q^T A-fragments)
//      ldsVt[64 dd ][32 keys] (transposed, for O^T = V^T @ P^T A-fragments)
//    P^T B-fragment is assembled from the S^T C-fragments with two xor-16
//    lane shuffles; softmax state stays lane-uniform (query = lane column).
// ---------------------------------------------------------------------------
__global__ __launch_bounds__(128)
void flash_attn_kernel(const _Float16* __restrict__ qkv,
                       _Float16* __restrict__ attnout) {
    __shared__ _Float16 ldsK[32 * HD];            // 4 KB
    __shared__ _Float16 ldsVt[HD * 32];           // 4 KB

    const int tid  = threadIdx.x;
    const int wave = tid >> 5;
    const int lane = tid & 31;
    const int l    = lane & 15;
    const int hi   = lane >> 4;

    const int bh   = blockIdx.x >> 5;             // 0..31 (32 q-blocks per bh)
    const int qblk = blockIdx.x & 31;
    const int b    = bh >> 4;
    const int h    = bh & 15;
    const int q0   = (qblk * 4 + wave) * 16;

    const size_t mstride  = (size_t)BATCH * NHEAD * SEQ * HD;
    const size_t head_off = ((size_t)b * NHEAD + h) * SEQ * HD;
    const _Float16* Q = qkv + head_off;
    const _Float16* K = qkv + mstride + head_off;
    const _Float16* V = qkv + 2 * mstride + head_off;

    // Q^T B-fragments are loop-invariant: B[k=dd][n=query] = Q[query][dd].
    const v16h bq0 = load_b_frag(Q, HD, q0 + l, 0,  hi);
    const v16h bq1 = load_b_frag(Q, HD, q0 + l, 32, hi);

    // cooperative staging pattern: 128 threads x 16 halves = 32x64 tile
    const int frow = tid >> 2;                    // 0..31 (key row)
    const int fcol = (tid & 3) * 16;              // 0,16,32,48 (dd)

    float m_run = -1e30f;
    float l_run = 0.0f;
    v8f oacc[4] = {{}, {}, {}, {}};               // O^T dim-tiles [dim, query]
    const float scale = 0.125f;                   // hd^-0.5

    for (int kk = 0; kk < SEQ; kk += 32) {
        __syncthreads();                          // LDS reuse from prev iter
        v16h kv = *(const v16h*)(K + (size_t)(kk + frow) * HD + fcol);
        v16h vv = *(const v16h*)(V + (size_t)(kk + frow) * HD + fcol);
        *(v16h*)(ldsK + frow * HD + fcol) = kv;   // row-major
#pragma unroll
        for (int ii = 0; ii < 16; ii++)           // transpose into ldsVt
            ldsVt[(fcol + ii) * 32 + frow] = vv[ii];
        if (kk + 32 < SEQ) {                      // prefetch next K/V tile
            __builtin_prefetch(K + (size_t)(kk + 32 + frow) * HD + fcol, 0, 3);
            __builtin_prefetch(V + (size_t)(kk + 32 + frow) * HD + fcol, 0, 3);
        }
        __syncthreads();

        // --- S^T tiles: keys kk..kk+15 (c0) and kk+16..kk+31 (c1) ----------
        v8f c0 = {}, c1 = {};
        c0 = wmma_f16(load_a_frag(ldsK, HD, l,      0,  hi), bq0, c0);
        c0 = wmma_f16(load_a_frag(ldsK, HD, l,      32, hi), bq1, c0);
        c1 = wmma_f16(load_a_frag(ldsK, HD, 16 + l, 0,  hi), bq0, c1);
        c1 = wmma_f16(load_a_frag(ldsK, HD, 16 + l, 32, hi), bq1, c1);

        // --- online softmax (per lane: 16 key-scores for query column l) --
        float s0[8], s1[8];
        float mt = -1e30f;
#pragma unroll
        for (int v = 0; v < 8; v++) {
            s0[v] = c0[v] * scale;
            s1[v] = c1[v] * scale;
            mt = fmaxf(mt, fmaxf(s0[v], s1[v]));
        }
        mt = fmaxf(mt, __shfl_xor(mt, 16, 32));
        const float m_new = fmaxf(m_run, mt);
        const float alpha = __expf(m_run - m_new);

        float p0[8], p1[8], ls = 0.0f;
#pragma unroll
        for (int v = 0; v < 8; v++) {
            p0[v] = __expf(s0[v] - m_new);
            p1[v] = __expf(s1[v] - m_new);
            ls += p0[v] + p1[v];
        }
        ls += __shfl_xor(ls, 16, 32);
        l_run = l_run * alpha + ls;
        m_run = m_new;
#pragma unroll
        for (int t = 0; t < 4; t++)
#pragma unroll
            for (int v = 0; v < 8; v++) oacc[t][v] *= alpha;

        // --- build P^T B-fragment via two xor-16 shuffles ------------------
        v8h ph0, ph1;
#pragma unroll
        for (int v = 0; v < 8; v++) {
            ph0[v] = (_Float16)p0[v];
            ph1[v] = (_Float16)p1[v];
        }
        v4i i0 = __builtin_bit_cast(v4i, ph0);
        v4i i1 = __builtin_bit_cast(v4i, ph1);
        v4i x0, x1;
#pragma unroll
        for (int v = 0; v < 4; v++) {
            x0[v] = __shfl_xor(i0[v], 16, 32);
            x1[v] = __shfl_xor(i1[v], 16, 32);
        }
        v8h oth0 = __builtin_bit_cast(v8h, x0);
        v8h oth1 = __builtin_bit_cast(v8h, x1);
        v16h bp_lo = __builtin_shufflevector(ph0, oth0,
                        0,1,2,3,4,5,6,7,8,9,10,11,12,13,14,15);
        v16h bp_hi = __builtin_shufflevector(oth1, ph1,
                        0,1,2,3,4,5,6,7,8,9,10,11,12,13,14,15);
        v16h bp = (hi == 0) ? bp_lo : bp_hi;

        // --- O^T += V^T @ P^T, V^T A-frags straight from transposed LDS ---
#pragma unroll
        for (int t = 0; t < 4; t++) {
            v16h av = load_a_frag(ldsVt, 32, t * 16 + l, 0, hi);
            oacc[t] = wmma_f16(av, bp, oacc[t]);
        }
    }

    // --- finalize + store O^T[dim, query] into [token][feature] f16 -------
    const float inv = 1.0f / l_run;
    const int token = b * SEQ + q0 + l;           // query column l
#pragma unroll
    for (int t = 0; t < 4; t++) {
        v8h oh;
#pragma unroll
        for (int v = 0; v < 8; v++)
            oh[v] = (_Float16)(oacc[t][v] * inv);
        const size_t off = (size_t)token * D_MODEL + h * HD + t * 16 + hi * 8;
        *(v8h*)(attnout + off) = oh;              // 16B contiguous store
    }
}

// ---------------------------------------------------------------------------
// 4) Output projection: [4096 x 1024] x Wo^T -> f32 d_out (64x64 per wave)
// ---------------------------------------------------------------------------
__global__ __launch_bounds__(128)
void out_gemm_kernel(const _Float16* __restrict__ ah,
                     const _Float16* __restrict__ wo,
                     float* __restrict__ out) {
    const int wave = threadIdx.x >> 5;
    const int lane = threadIdx.x & 31;
    const int l  = lane & 15;
    const int hi = lane >> 4;

    const int NT   = D_MODEL / 64;                // 16 col tiles
    const int tile = blockIdx.x * 4 + wave;       // 0..1023
    const int tm   = tile / NT;
    const int tn   = tile % NT;
    const int row0 = tm * 64;
    const int col0 = tn * 64;

    v8f c[4][4] = {};
    for (int k = 0; k < D_MODEL; k += 32) {
        v16h a[4], b[4];
#pragma unroll
        for (int i = 0; i < 4; i++)
            a[i] = load_a_frag(ah, D_MODEL, row0 + i * 16 + l, k, hi);
#pragma unroll
        for (int j = 0; j < 4; j++)
            b[j] = load_b_frag(wo, D_MODEL, col0 + j * 16 + l, k, hi);
        if (k + 128 < D_MODEL) {
            __builtin_prefetch(ah + (size_t)(row0 + l) * D_MODEL + k + 128, 0, 3);
            __builtin_prefetch(wo + (size_t)(col0 + l) * D_MODEL + k + 128, 0, 3);
        }
#pragma unroll
        for (int i = 0; i < 4; i++)
#pragma unroll
            for (int j = 0; j < 4; j++)
                c[i][j] = wmma_f16(a[i], b[j], c[i][j]);
    }
#pragma unroll
    for (int i = 0; i < 4; i++)
#pragma unroll
        for (int j = 0; j < 4; j++)
#pragma unroll
            for (int v = 0; v < 8; v++)
                out[(size_t)(row0 + i * 16 + v + 8 * hi) * D_MODEL
                    + col0 + j * 16 + l] = c[i][j][v];
}

// ---------------------------------------------------------------------------
// Launch
// ---------------------------------------------------------------------------
extern "C" void kernel_launch(void* const* d_in, const int* in_sizes, int n_in,
                              void* d_out, int out_size, void* d_ws, size_t ws_size,
                              hipStream_t stream) {
    (void)in_sizes; (void)n_in; (void)out_size; (void)ws_size;

    const float* x  = (const float*)d_in[0];
    const float* Wq = (const float*)d_in[1];
    const float* Wk = (const float*)d_in[2];
    const float* Wv = (const float*)d_in[3];
    const float* Wo = (const float*)d_in[4];
    float* out = (float*)d_out;

    // Workspace layout (f16), total 48 MB:
    char* ws = (char*)d_ws;
    const size_t XH_BYTES = (size_t)TOKENS * D_MODEL * 2;         // 8 MB
    const size_t W_BYTES  = (size_t)D_MODEL * D_MODEL * 2;        // 2 MB each
    _Float16* xh   = (_Float16*)(ws);
    _Float16* wqh  = (_Float16*)(ws + XH_BYTES);
    _Float16* wkh  = (_Float16*)(ws + XH_BYTES + W_BYTES);
    _Float16* wvh  = (_Float16*)(ws + XH_BYTES + 2 * W_BYTES);
    _Float16* woh  = (_Float16*)(ws + XH_BYTES + 3 * W_BYTES);
    _Float16* qkv  = (_Float16*)(ws + XH_BYTES + 4 * W_BYTES);    // 24 MB
    _Float16* attn = (_Float16*)(ws + XH_BYTES + 4 * W_BYTES + 3 * XH_BYTES);

    const int n_x = TOKENS * D_MODEL;       // 4,194,304
    const int n_w = D_MODEL * D_MODEL;      // 1,048,576

    f32_to_f16_kernel<<<4096, 256, 0, stream>>>(x,  xh,  n_x);
    f32_to_f16_kernel<<<2048, 256, 0, stream>>>(Wq, wqh, n_w);
    f32_to_f16_kernel<<<2048, 256, 0, stream>>>(Wk, wkh, n_w);
    f32_to_f16_kernel<<<2048, 256, 0, stream>>>(Wv, wvh, n_w);
    f32_to_f16_kernel<<<2048, 256, 0, stream>>>(Wo, woh, n_w);

    // QKV: 64 row-tiles * 48 col-tiles = 3072 waves / 4 per block
    qkv_gemm_kernel<<<768, 128, 0, stream>>>(xh, wqh, wkh, wvh, qkv);

    // Attention: 32 (b,h) * 32 query blocks
    flash_attn_kernel<<<1024, 128, 0, stream>>>(qkv, attn);

    // Output projection: 64 * 16 = 1024 waves / 4 per block
    out_gemm_kernel<<<256, 128, 0, stream>>>(attn, woh, out);
}
